// RNNModel_87660282511993
// MI455X (gfx1250) — matrix operand order repaired and checked by reading, not compiled, not run on hardware
//
#include <hip/hip_runtime.h>
#include <math.h>

// ---------------- model dims ----------------
#define VOCAB   30000
#define POSD    8
#define NNL1    25
#define NNL2    10
#define BB      32
#define LL      100
#define ROWS    (BB*LL)            // 3200
#define R2      (ROWS*NNL1)        // 80000
#define KDEC    128                // 2*H
#define NT16    (VOCAB/16)         // 1875 16-col subtiles
#define NG4     ((NT16 + 3)/4)     // 469 groups of 4 subtiles (last ragged)

typedef float     v8f  __attribute__((ext_vector_type(8)));
typedef float     v2f  __attribute__((ext_vector_type(2)));
typedef _Float16  v16h __attribute__((ext_vector_type(16)));

union frag16 { v16h v; float4 f4[2]; };

static __device__ __forceinline__ v8f v8zero() {
    v8f v;
#pragma unroll
    for (int i = 0; i < 8; ++i) v[i] = 0.f;
    return v;
}

static __device__ __forceinline__ float sigmoidf_(float x) {
    return 1.f / (1.f + expf(-x));
}

// ------------------------------------------------------------------
// Generic fp32 WMMA GEMM:  C[M,N] = act(A[M,K] @ W[N,K]^T + bias[N])
// One wave = one 16x16 tile via V_WMMA_F32_16X16X4_F32 chain (exact fp32).
// Used for the small GCN / GRU-input GEMMs.
// ------------------------------------------------------------------
__global__ __launch_bounds__(256) void gemm_wmma_f32(
    const float* __restrict__ A, const float* __restrict__ W,
    const float* __restrict__ bias, float* __restrict__ C,
    int M, int N, int K, int ldc, int relu)
{
    int wid  = (blockIdx.x * blockDim.x + threadIdx.x) >> 5;
    int lane = threadIdx.x & 31;
    int ntn  = N >> 4;
    int total = (M >> 4) * ntn;
    if (wid >= total) return;                 // uniform per wave
    int tm = wid / ntn, tn = wid % ntn;
    int l15  = lane & 15;
    int half = lane >> 4;                     // 0: K=0,1   1: K=2,3

    const float* arow = A + (size_t)(tm * 16 + l15) * K;
    const float* wrow = W + (size_t)(tn * 16 + l15) * K;

    v8f acc = v8zero();
    for (int k = 0; k < K; k += 4) {
        float4 av = *(const float4*)(arow + k);
        float4 wv = *(const float4*)(wrow + k);
        v2f af, bf;
        af.x = half ? av.z : av.x;  af.y = half ? av.w : av.y;
        bf.x = half ? wv.z : wv.x;  bf.y = half ? wv.w : wv.y;
        acc = __builtin_amdgcn_wmma_f32_16x16x4_f32(
            false, af, false, bf, (short)0, acc, false, false);
    }

    float bcol = bias[tn * 16 + l15];
    float* ccol = C + (tn * 16 + l15);
#pragma unroll
    for (int i = 0; i < 8; ++i) {
        int m = tm * 16 + i + half * 8;
        float v = acc[i] + bcol;
        if (relu) v = fmaxf(v, 0.f);
        ccol[(size_t)m * ldc] = v;
    }
}

// ------------------------------------------------------------------
// Decoder GEMM, f16 inputs / f32 accumulate, 64x64 register-blocked
// wave tiles: 4x4 subtiles of V_WMMA_F32_16X16X32_F16, K=128 in 4 steps.
// Ragged N edge handled by clamping load rows (never branching around
// WMMA -> EXEC stays all-1s) and predicating only the epilogue stores.
// ------------------------------------------------------------------
__global__ __launch_bounds__(256) void dec_gemm_f16(
    const _Float16* __restrict__ Ah,   // [ROWS][128]
    const _Float16* __restrict__ Wh,   // [VOCAB][128]
    const float* __restrict__ bias,    // [VOCAB]
    float* __restrict__ C)             // [ROWS][VOCAB]
{
    int wid  = (blockIdx.x * blockDim.x + threadIdx.x) >> 5;
    int lane = threadIdx.x & 31;
    int total = (ROWS / 64) * NG4;               // 50 * 469
    if (wid >= total) return;
    int tm = wid / NG4;                          // 64-row tile index
    int tg = wid % NG4;                          // group of 4 N-subtiles
    int l15 = lane & 15, half = lane >> 4;
    int nsub0 = tg * 4;

    const _Float16* arow[4];
#pragma unroll
    for (int mi = 0; mi < 4; ++mi)
        arow[mi] = Ah + (size_t)(tm * 64 + mi * 16 + l15) * KDEC;

    const _Float16* wrow[4];
#pragma unroll
    for (int ni = 0; ni < 4; ++ni) {
        int nr = (nsub0 + ni) * 16 + l15;
        if (nr > VOCAB - 1) nr = VOCAB - 1;      // clamp: garbage math, never stored
        wrow[ni] = Wh + (size_t)nr * KDEC;
    }

    v8f acc[4][4];
#pragma unroll
    for (int mi = 0; mi < 4; ++mi)
#pragma unroll
        for (int ni = 0; ni < 4; ++ni) acc[mi][ni] = v8zero();

#pragma unroll
    for (int k = 0; k < KDEC; k += 32) {
        frag16 a[4], b[4];
#pragma unroll
        for (int mi = 0; mi < 4; ++mi) {
            a[mi].f4[0] = *(const float4*)(arow[mi] + k + half * 8);
            a[mi].f4[1] = *(const float4*)(arow[mi] + k + half * 8 + 16);
        }
#pragma unroll
        for (int ni = 0; ni < 4; ++ni) {
            b[ni].f4[0] = *(const float4*)(wrow[ni] + k + half * 8);
            b[ni].f4[1] = *(const float4*)(wrow[ni] + k + half * 8 + 16);
        }
#pragma unroll
        for (int mi = 0; mi < 4; ++mi)
#pragma unroll
            for (int ni = 0; ni < 4; ++ni)
                acc[mi][ni] = __builtin_amdgcn_wmma_f32_16x16x32_f16(
                    false, a[mi].v, false, b[ni].v, (short)0, acc[mi][ni], false, false);
    }

#pragma unroll
    for (int ni = 0; ni < 4; ++ni) {
        int col = (nsub0 + ni) * 16 + l15;
        bool ok = col < VOCAB;                   // store predication only
        float bc = bias[ok ? col : 0];
#pragma unroll
        for (int mi = 0; mi < 4; ++mi)
#pragma unroll
            for (int i = 0; i < 8; ++i) {
                if (ok)
                    C[(size_t)(tm * 64 + mi * 16 + i + half * 8) * VOCAB + col]
                        = acc[mi][ni][i] + bc;
            }
    }
}

// ---------------- f32 -> f16 convert ----------------
__global__ void f32_to_f16(const float* __restrict__ in,
                           _Float16* __restrict__ out, int n)
{
    int i = blockIdx.x * blockDim.x + threadIdx.x;
    if (i < n) out[i] = (_Float16)in[i];
}

// ---------------- gather + mean over NNL2 neighbors ----------------
__global__ void gather_mean10(const float* __restrict__ tab,
                              const int* __restrict__ nb2,
                              float* __restrict__ out)  // [R2][64]
{
    int idx = blockIdx.x * blockDim.x + threadIdx.x;
    if (idx >= R2 * 64) return;
    int r = idx >> 6, c = idx & 63;
    const int* ind = nb2 + r * NNL2;
    float s = 0.f;
#pragma unroll
    for (int j = 0; j < NNL2; ++j)
        s += tab[(size_t)ind[j] * 64 + c];
    out[idx] = s * (1.f / NNL2);
}

// ---------------- mean over NNL1 group ----------------
__global__ void mean25(const float* __restrict__ in,    // [R2][64]
                       float* __restrict__ out)         // [ROWS][64]
{
    int idx = blockIdx.x * blockDim.x + threadIdx.x;
    if (idx >= ROWS * 64) return;
    int g = idx >> 6, c = idx & 63;
    float s = 0.f;
#pragma unroll
    for (int i = 0; i < NNL1; ++i)
        s += in[(size_t)(g * NNL1 + i) * 64 + c];
    out[idx] = s * (1.f / NNL1);
}

// ---- net_mem sliding window -> opack_h[:,64:128] (f16) ----
__global__ void netmem_pack(const float* __restrict__ nf1,  // [ROWS][64]
                            _Float16* __restrict__ opack_h) // [ROWS][128]
{
    int idx = blockIdx.x * blockDim.x + threadIdx.x;
    if (idx >= ROWS * 64) return;
    int r = idx >> 6, c = idx & 63;
    int t = r % LL;
    float s = nf1[(size_t)r * 64 + c];
    if (t >= 1) s += nf1[(size_t)(r - 1) * 64 + c];
    if (t >= 2) s += nf1[(size_t)(r - 2) * 64 + c];
    opack_h[(size_t)r * 128 + 64 + c] = (_Float16)(s * (1.f / 3.f));
}

// ---- build x = [encoder_w[inp], pos] : [ROWS][72] ----
__global__ void build_x(const float* __restrict__ enc,
                        const float* __restrict__ pos,
                        const int* __restrict__ seq,
                        float* __restrict__ x)
{
    int idx = blockIdx.x * blockDim.x + threadIdx.x;
    if (idx >= ROWS * 72) return;
    int r = idx / 72, c = idx % 72;
    int b = r / LL, t = r % LL;
    float v;
    if (c < 64) v = enc[(size_t)seq[b * (LL + 1) + t] * 64 + c];
    else        v = pos[t * POSD + (c - 64)];
    x[idx] = v;
}

// ------------------------------------------------------------------
// GRU: 1 block, 256 threads (8 waves). W_hh B-fragments persistent in
// registers; h staged in padded LDS; gh via 48 fp32 WMMA per wave/step.
// ------------------------------------------------------------------
__global__ __launch_bounds__(256) void gru_kernel(
    const float* __restrict__ gi_all,  // [ROWS][192] (includes b_ih)
    const float* __restrict__ w_hh,    // [192][64]
    const float* __restrict__ b_hh,    // [192]
    _Float16* __restrict__ opack_h,    // [ROWS][128]  (writes cols 0..63)
    float* __restrict__ hidden_out)    // [32][64]
{
    __shared__ __align__(16) float h_lds[2][32][68];   // pad 68: no bank conflicts
    __shared__ float gh_lds[32][196];                  // pad 196

    int tid  = threadIdx.x;
    int lane = tid & 31, wid = tid >> 5;
    int l15  = lane & 15, half = lane >> 4;
    int mtile = wid & 1;
    int n0    = (wid >> 1) * 3;                        // 3 N-tiles per wave

    for (int i = tid; i < 32 * 68; i += 256)
        ((float*)h_lds[0])[i] = 0.f;
    __syncthreads();

    v2f bfrag[3][16];
#pragma unroll
    for (int nt = 0; nt < 3; ++nt) {
        const float* wr = w_hh + (size_t)((n0 + nt) * 16 + l15) * 64;
#pragma unroll
        for (int ks = 0; ks < 16; ++ks) {
            bfrag[nt][ks].x = wr[ks * 4 + half * 2];
            bfrag[nt][ks].y = wr[ks * 4 + half * 2 + 1];
        }
    }

    int cur = 0;
    for (int t = 0; t < LL; ++t) {
        v8f a0 = v8zero(), a1 = v8zero(), a2 = v8zero();
#pragma unroll
        for (int ks = 0; ks < 16; ++ks) {
            float4 hv = *(const float4*)&h_lds[cur][mtile * 16 + l15][ks * 4];
            v2f af;
            af.x = half ? hv.z : hv.x;  af.y = half ? hv.w : hv.y;
            a0 = __builtin_amdgcn_wmma_f32_16x16x4_f32(false, af, false, bfrag[0][ks], (short)0, a0, false, false);
            a1 = __builtin_amdgcn_wmma_f32_16x16x4_f32(false, af, false, bfrag[1][ks], (short)0, a1, false, false);
            a2 = __builtin_amdgcn_wmma_f32_16x16x4_f32(false, af, false, bfrag[2][ks], (short)0, a2, false, false);
        }
#pragma unroll
        for (int i = 0; i < 8; ++i) {
            int m = mtile * 16 + i + half * 8;
            gh_lds[m][(n0 + 0) * 16 + l15] = a0[i];
            gh_lds[m][(n0 + 1) * 16 + l15] = a1[i];
            gh_lds[m][(n0 + 2) * 16 + l15] = a2[i];
        }
        __syncthreads();

        for (int e = tid; e < 32 * 64; e += 256) {
            int i = e >> 6, j = e & 63;
            const float* gir = gi_all + (size_t)(i * LL + t) * 192;
            float r = sigmoidf_(gir[j]       + gh_lds[i][j]       + b_hh[j]);
            float z = sigmoidf_(gir[64 + j]  + gh_lds[i][64 + j]  + b_hh[64 + j]);
            float n = tanhf(gir[128 + j] + r * (gh_lds[i][128 + j] + b_hh[128 + j]));
            float ho = h_lds[cur][i][j];
            float hn = (1.f - z) * n + z * ho;
            h_lds[cur ^ 1][i][j] = hn;
            opack_h[(size_t)(i * LL + t) * 128 + j] = (_Float16)hn;
        }
        __syncthreads();
        cur ^= 1;
    }

    for (int e = tid; e < 32 * 64; e += 256)
        hidden_out[e] = h_lds[cur][e >> 6][e & 63];
}

// ---- causal -inf mask scatter (set == add(-inf) here) ----
__global__ void mask_kernel(const int* __restrict__ seq,
                            float* __restrict__ out)
{
    int id = blockIdx.x * blockDim.x + threadIdx.x;   // b*LL + j
    if (id >= ROWS) return;
    int b = id / LL, j = id % LL;
    const float NEG = -__builtin_inff();
    out[(size_t)id * VOCAB] = NEG;                    // vocab index 0 always masked
    int tok = seq[b * (LL + 1) + j];
    for (int t = j; t < LL; ++t)
        out[(size_t)(b * LL + t) * VOCAB + tok] = NEG;
}

// ------------------------------------------------------------------
extern "C" void kernel_launch(void* const* d_in, const int* in_sizes, int n_in,
                              void* d_out, int out_size, void* d_ws, size_t ws_size,
                              hipStream_t stream) {
    const float* encoder_w = (const float*)d_in[0];
    const float* pos_w     = (const float*)d_in[1];
    const float* w_ih      = (const float*)d_in[2];
    const float* w_hh      = (const float*)d_in[3];
    const float* b_ih      = (const float*)d_in[4];
    const float* b_hh      = (const float*)d_in[5];
    const float* gcn1_w    = (const float*)d_in[6];
    const float* gcn1_b    = (const float*)d_in[7];
    const float* gcn2_w    = (const float*)d_in[8];
    const float* gcn2_b    = (const float*)d_in[9];
    const float* dec_w     = (const float*)d_in[10];
    const float* dec_b     = (const float*)d_in[11];
    const float* net_tab   = (const float*)d_in[12];
    const int*   seq       = (const int*)d_in[13];
    const int*   nb2       = (const int*)d_in[14];

    float* result = (float*)d_out;                          // [ROWS][VOCAB]
    float* hidden = (float*)d_out + (size_t)ROWS * VOCAB;   // [32][64]

    // workspace layout (float units; all offsets 16B-aligned)
    float* ws     = (float*)d_ws;
    float* Mfeat  = ws;                           // R2*64
    float* nf2    = Mfeat  + (size_t)R2 * 64;     // R2*64
    float* nf2m   = nf2    + (size_t)R2 * 64;     // ROWS*64
    float* nf1    = nf2m   + (size_t)ROWS * 64;   // ROWS*64
    float* xbuf   = nf1    + (size_t)ROWS * 64;   // ROWS*72
    float* gi_all = xbuf   + (size_t)ROWS * 72;   // ROWS*192
    _Float16* dec_h   = (_Float16*)(gi_all + (size_t)ROWS * 192); // VOCAB*128 halves
    _Float16* opack_h = dec_h + (size_t)VOCAB * KDEC;             // ROWS*128 halves

    auto gemm = [&](const float* A, const float* W, const float* B, float* C,
                    int M, int N, int K, int ldc, int relu) {
        int waves  = (M / 16) * (N / 16);
        int blocks = (waves + 7) / 8;
        gemm_wmma_f32<<<blocks, 256, 0, stream>>>(A, W, B, C, M, N, K, ldc, relu);
    };

    // 0) convert dec_w to f16 (independent of everything else)
    f32_to_f16<<<(VOCAB * KDEC + 255) / 256, 256, 0, stream>>>(dec_w, dec_h, VOCAB * KDEC);
    // 1) gather + mean over 10 neighbors
    gather_mean10<<<(R2 * 64 + 255) / 256, 256, 0, stream>>>(net_tab, nb2, Mfeat);
    // 2) nf2 = relu(Mfeat @ gcn2^T + b2)
    gemm(Mfeat, gcn2_w, gcn2_b, nf2, R2, 64, 64, 64, 1);
    // 3) mean over 25
    mean25<<<(ROWS * 64 + 255) / 256, 256, 0, stream>>>(nf2, nf2m);
    // 4) nf1 = relu(nf2m @ gcn1^T + b1)
    gemm(nf2m, gcn1_w, gcn1_b, nf1, ROWS, 64, 64, 64, 1);
    // 5) net_mem sliding window -> opack_h[:,64:128]
    netmem_pack<<<(ROWS * 64 + 255) / 256, 256, 0, stream>>>(nf1, opack_h);
    // 6) x = [emb, pos]
    build_x<<<(ROWS * 72 + 255) / 256, 256, 0, stream>>>(encoder_w, pos_w, seq, xbuf);
    // 7) gi_all = x @ w_ih^T + b_ih  (no recurrence -> fully parallel)
    gemm(xbuf, w_ih, b_ih, gi_all, ROWS, 192, 72, 192, 0);
    // 8) GRU recurrence -> opack_h[:,0:64] and hidden
    gru_kernel<<<1, 256, 0, stream>>>(gi_all, w_hh, b_hh, opack_h, hidden);
    // 9) decoder: result = opack_h @ dec_h^T + dec_b  (f16 WMMA, 64x64 wave tiles)
    {
        int waves  = (ROWS / 64) * NG4;           // 50 * 469 = 23450
        int blocks = (waves + 7) / 8;
        dec_gemm_f16<<<blocks, 256, 0, stream>>>(opack_h, dec_h, dec_b, result);
    }
    // 10) causal -inf mask
    mask_kernel<<<(ROWS + 255) / 256, 256, 0, stream>>>(seq, result);
}
